// Int8Linear_24146306138245
// MI455X (gfx1250) — 8-bit matrix operand order repaired and checked by reading, compiled, not run on hardware
//
#include <hip/hip_runtime.h>

typedef __attribute__((ext_vector_type(8))) int v8i;
typedef __attribute__((ext_vector_type(4))) int v4i;
typedef __attribute__((ext_vector_type(2))) int v2i;

#define BM 128
#define BN 128
#define BK 64
#define LSTR 80

#if defined(__has_builtin)
#if __has_builtin(__builtin_amdgcn_global_load_async_to_lds_b128) && \
    __has_builtin(__builtin_amdgcn_s_wait_asynccnt)
#define USE_ASYNC_LDS 1
#endif
#endif
#ifndef USE_ASYNC_LDS
#define USE_ASYNC_LDS 0
#endif

#if USE_ASYNC_LDS
typedef __attribute__((address_space(1))) v4i gbl_v4i;
typedef __attribute__((address_space(3))) v4i lds_v4i;
#endif

__global__ void init_kernel(unsigned int* p) { *p = 0u; }

__global__ __launch_bounds__(256)
void absmax_kernel(const float* __restrict__ x, long long n4, unsigned int* __restrict__ smax) {
    __shared__ float red[256];
    float m = 0.0f;
    const float4* x4 = (const float4*)x;
    for (long long i = (long long)blockIdx.x * 256 + threadIdx.x; i < n4;
         i += 256ll * gridDim.x) {
        float4 v = x4[i];
        m = fmaxf(m, fmaxf(fmaxf(fabsf(v.x), fabsf(v.y)),
                           fmaxf(fabsf(v.z), fabsf(v.w))));
    }
    red[threadIdx.x] = m;
    __syncthreads();
    for (int s = 128; s > 0; s >>= 1) {
        if ((int)threadIdx.x < s)
            red[threadIdx.x] = fmaxf(red[threadIdx.x], red[threadIdx.x + s]);
        __syncthreads();
    }
    if (threadIdx.x == 0)
        atomicMax(smax, __float_as_uint(red[0]));
}

__global__ __launch_bounds__(256)
void quantize_kernel(const float* __restrict__ x, int* __restrict__ xq4,
                     const unsigned int* __restrict__ smax, long long n4) {
    long long i = (long long)blockIdx.x * 256 + threadIdx.x;
    if (i >= n4) return;
    const float s = __uint_as_float(*smax) / 127.0f;
    float4 v = ((const float4*)x)[i];
    int b0 = (int)fminf(fmaxf(v.x / s, -127.0f), 127.0f);
    int b1 = (int)fminf(fmaxf(v.y / s, -127.0f), 127.0f);
    int b2 = (int)fminf(fmaxf(v.z / s, -127.0f), 127.0f);
    int b3 = (int)fminf(fmaxf(v.w / s, -127.0f), 127.0f);
    xq4[i] = (b0 & 255) | ((b1 & 255) << 8) | ((b2 & 255) << 16) | ((b3 & 255) << 24);
}

__global__ __launch_bounds__(256)
void gemm_i8_kernel(const signed char* __restrict__ xq,
                    const signed char* __restrict__ w,
                    float* __restrict__ out,
                    const unsigned int* __restrict__ smax,
                    const float* __restrict__ wscale,
                    const float* __restrict__ bias,
                    int N, int K) {
    __shared__ __align__(16) signed char As[BM * LSTR];
    __shared__ __align__(16) signed char Bs[BN * LSTR];

    const int tid  = threadIdx.x;
    const int lane = tid & 31;
    const int wave = tid >> 5;
    const int wm   = wave & 3;
    const int wn   = wave >> 2;
    const int nl   = lane & 15;
    const int hi   = lane >> 4;

    const long long rowA0 = (long long)blockIdx.y * BM;
    const long long rowB0 = (long long)blockIdx.x * BN;

    v8i acc[2][4];
#pragma unroll
    for (int i = 0; i < 2; ++i)
#pragma unroll
        for (int j = 0; j < 4; ++j)
            acc[i][j] = (v8i){0, 0, 0, 0, 0, 0, 0, 0};

    const int kTiles = K / BK;
    for (int kt = 0; kt < kTiles; ++kt) {
        __syncthreads();
#pragma unroll
        for (int it = 0; it < 2; ++it) {
            const int c   = tid + it * 256;
            const int row = c >> 2;
            const int col = (c & 3) << 4;
            const long long gk = (long long)kt * BK + col;
            const signed char* ga = xq + (rowA0 + row) * K + gk;
            const signed char* gb = w  + (rowB0 + row) * K + gk;
#if USE_ASYNC_LDS
            __builtin_amdgcn_global_load_async_to_lds_b128(
                (gbl_v4i*)ga, (lds_v4i*)(As + row * LSTR + col), 0, 0);
            __builtin_amdgcn_global_load_async_to_lds_b128(
                (gbl_v4i*)gb, (lds_v4i*)(Bs + row * LSTR + col), 0, 0);
#else
            v4i a = *(const v4i*)ga;
            v4i b = *(const v4i*)gb;
            *(v4i*)(As + row * LSTR + col) = a;
            *(v4i*)(Bs + row * LSTR + col) = b;
#endif
        }
#if USE_ASYNC_LDS
        __builtin_amdgcn_s_wait_asynccnt(0);
#endif
        __syncthreads();

        v8i afrag[2];
#pragma unroll
        for (int tm = 0; tm < 2; ++tm) {
            const int r    = wm * 32 + tm * 16 + nl;
            const int base = r * LSTR + hi * 16;
            v2i q0 = *(const v2i*)(As + base);
            v2i q1 = *(const v2i*)(As + base + 8);
            v2i q2 = *(const v2i*)(As + base + 32);
            v2i q3 = *(const v2i*)(As + base + 40);
            v8i a;
            a[0] = q0[0]; a[1] = q0[1]; a[2] = q1[0]; a[3] = q1[1];
            a[4] = q2[0]; a[5] = q2[1]; a[6] = q3[0]; a[7] = q3[1];
            afrag[tm] = a;
        }
        v8i bfrag[4];
#pragma unroll
        for (int tn = 0; tn < 4; ++tn) {
            const int r    = wn * 64 + tn * 16 + nl;
            const int base = r * LSTR + hi * 16;
            v4i q0 = *(const v4i*)(Bs + base);
            v4i q1 = *(const v4i*)(Bs + base + 32);
            v8i b;
            b[0] = q0[0]; b[1] = q0[1]; b[2] = q0[2]; b[3] = q0[3];
            b[4] = q1[0]; b[5] = q1[1]; b[6] = q1[2]; b[7] = q1[3];
            bfrag[tn] = b;
        }
#pragma unroll
        for (int tm = 0; tm < 2; ++tm)
#pragma unroll
            for (int tn = 0; tn < 4; ++tn)
                acc[tm][tn] = __builtin_amdgcn_wmma_i32_16x16x64_iu8(
                     true, afrag[tm],  true, bfrag[tn],
                    acc[tm][tn],  false,  false);
    }

    const float scale = (__uint_as_float(*smax) / 127.0f) * (*wscale);
#pragma unroll
    for (int tn = 0; tn < 4; ++tn) {
        const long long gn = rowB0 + wn * 64 + tn * 16 + nl;
        const float bv = bias[gn];
#pragma unroll
        for (int tm = 0; tm < 2; ++tm) {
            const long long gm0 = rowA0 + wm * 32 + tm * 16 + hi * 8;
#pragma unroll
            for (int v = 0; v < 8; ++v)
                out[(gm0 + v) * N + gn] = (float)acc[tm][tn][v] * scale + bv;
        }
    }
}

extern "C" void kernel_launch(void* const* d_in, const int* in_sizes, int n_in,
                              void* d_out, int out_size, void* d_ws, size_t ws_size,
                              hipStream_t stream) {
    const float*       x      = (const float*)d_in[0];
    const signed char* weight = (const signed char*)d_in[1];
    const float*       wscale = (const float*)d_in[2];
    const float*       bias   = (const float*)d_in[3];
    float*             out    = (float*)d_out;

    const int       N  = in_sizes[3];
    const long long NK = (long long)in_sizes[1];
    const int       K  = (int)(NK / N);
    const long long MK = (long long)in_sizes[0];
    const int       M  = (int)(MK / K);

    unsigned int* smax = (unsigned int*)d_ws;
    signed char*  xq   = (signed char*)d_ws + 256;

    const long long n4 = MK / 4;

    init_kernel<<<1, 1, 0, stream>>>(smax);
    absmax_kernel<<<2048, 256, 0, stream>>>(x, n4, smax);
    quantize_kernel<<<(unsigned)((n4 + 255) / 256), 256, 0, stream>>>(x, (int*)xq, smax, n4);

    dim3 grid(N / BN, M / BM);
    gemm_i8_kernel<<<grid, 256, 0, stream>>>(xq, weight, out, smax, wscale, bias, N, K);
}
